// AttentionOp_52432960749874
// MI455X (gfx1250) — matrix operand order repaired and checked by reading, not compiled, not run on hardware
//
#include <hip/hip_runtime.h>

typedef __attribute__((ext_vector_type(16))) __bf16 v16bf;
typedef __attribute__((ext_vector_type(8)))  float  v8f;

// Problem constants (match setup_inputs: B=4, S=4096, H=16, D=128, chunk=1024)
constexpr int B_ = 4, S_ = 4096, H_ = 16, D_ = 128, C_ = 1024;
constexpr int NCHUNK = S_ / C_;            // 4
constexpr int QT_ROWS = 128;               // query rows per workgroup (8 waves x 16)
constexpr int QT_PER_CHUNK = C_ / QT_ROWS; // 8
constexpr int KB = 32;                     // keys per inner iteration (WMMA K for PV)
constexpr int KPAD = 134;                  // K tile LDS row stride (bank-conflict pad)
constexpr int VPAD = 34;                   // V^T tile LDS row stride
constexpr float SCALE = 0.08838834764831845f; // 1/sqrt(128)

union BfVec {
    v16bf v;
    unsigned short us[16];
    unsigned int   u32[8];
};

__device__ __forceinline__ unsigned short f2bf(float f) {
    unsigned int u = __float_as_uint(f);
    u += 0x7FFFu + ((u >> 16) & 1u);       // round-to-nearest-even
    return (unsigned short)(u >> 16);
}
__device__ __forceinline__ unsigned int pack2bf(float a, float b) {
    return (unsigned int)f2bf(a) | ((unsigned int)f2bf(b) << 16);
}

__device__ __forceinline__ float half16_max(float x) {
#pragma unroll
    for (int m = 1; m < 16; m <<= 1) x = fmaxf(x, __shfl_xor(x, m, 32));
    return x;
}

__global__ __launch_bounds__(256, 1)
void attn_sink_kernel(const float* __restrict__ q, const float* __restrict__ k,
                      const float* __restrict__ v, const float* __restrict__ sinks,
                      float* __restrict__ out)
{
    __shared__ unsigned short Klds[KB][KPAD];     // K block, row-major (key, d), bf16
    __shared__ unsigned short Vtlds[D_][VPAD];    // V block, transposed (d, key), bf16
    __shared__ unsigned short Plds[8][16][KB];    // per-wave P scratch (row, key)

    int idx = blockIdx.x;
    const int qt = idx & (QT_PER_CHUNK - 1); idx >>= 3;
    const int n  = idx & (NCHUNK - 1);       idx >>= 2;
    const int h  = idx & (H_ - 1);           idx >>= 4;
    const int b  = idx;

    const int tid  = threadIdx.x;
    const int wave = tid >> 5;
    const int lane = tid & 31;
    const int lm   = lane & 15;
    const int half = lane >> 4;

    const int q0 = qt * QT_ROWS + wave * 16;      // within-chunk first query row for this wave

    const size_t rowStride = (size_t)H_ * D_;     // stride between consecutive s
    const float* qbase = q + (((size_t)b * S_ + (size_t)n * C_) * H_ + h) * D_;
    const float* kbase = k + (((size_t)b * S_ + (size_t)n * C_) * H_ + h) * D_;
    const float* vbase = v + (((size_t)b * S_ + (size_t)n * C_) * H_ + h) * D_;

    // ---- Preload Q tile (16 rows x 128 d), 1/sqrt(D) folded in, A-layout bf16 ----
    BfVec aq[4];
    {
        const float* qp = qbase + (size_t)(q0 + lm) * rowStride;
#pragma unroll
        for (int ds = 0; ds < 4; ++ds) {
#pragma unroll
            for (int p = 0; p < 8; ++p) {
                const int d = ds * 32 + 16 * (p >> 2) + 8 * half + 2 * (p & 3);
                aq[ds].u32[p] = pack2bf(qp[d] * SCALE, qp[d + 1] * SCALE);
            }
        }
    }

    // all-ones 32x16 bf16 B operand: row-sum of P via the matrix core
    BfVec onesB;
#pragma unroll
    for (int i = 0; i < 16; ++i) onesB.us[i] = 0x3F80;   // bf16 1.0

    // ---- Online softmax state; sink folds in as initial (m = sink, l = 1) ----
    const float sink = sinks[h];
    float m_i[8], l_i[8];
    v8f acc[8];
#pragma unroll
    for (int r = 0; r < 8; ++r) { m_i[r] = sink; l_i[r] = 1.0f; }
#pragma unroll
    for (int t = 0; t < 8; ++t)
#pragma unroll
        for (int e = 0; e < 8; ++e) acc[t][e] = 0.0f;

    const int kbN = (qt + 1) * (QT_ROWS / KB);    // causal bound for this workgroup

    // ---- software pipeline: this thread's 16 K-floats + 16 V-floats per block ----
    const int skey = tid >> 3;                    // 0..31
    const int sd0  = (tid & 7) * 16;              // 0..112
    float4 kreg[4], vreg[4];
    auto loadKV = [&](int kb) {
        const float4* kp = reinterpret_cast<const float4*>(
            kbase + (size_t)(kb * KB + skey) * rowStride + sd0);
        const float4* vp = reinterpret_cast<const float4*>(
            vbase + (size_t)(kb * KB + skey) * rowStride + sd0);
#pragma unroll
        for (int i = 0; i < 4; ++i) { kreg[i] = kp[i]; vreg[i] = vp[i]; }
    };
    loadKV(0);

    for (int kb = 0; kb < kbN; ++kb) {
        // ---- Stage current block (registers -> LDS, f32 -> bf16) ----
        {
#pragma unroll
            for (int i = 0; i < 4; ++i) {
                const int d = sd0 + 4 * i;
                *(unsigned int*)&Klds[skey][d]     = pack2bf(kreg[i].x, kreg[i].y);
                *(unsigned int*)&Klds[skey][d + 2] = pack2bf(kreg[i].z, kreg[i].w);
                Vtlds[d + 0][skey] = f2bf(vreg[i].x);
                Vtlds[d + 1][skey] = f2bf(vreg[i].y);
                Vtlds[d + 2][skey] = f2bf(vreg[i].z);
                Vtlds[d + 3][skey] = f2bf(vreg[i].w);
            }
        }
        __syncthreads();

        // issue next block's global loads; they overlap this block's compute
        if (kb + 1 < kbN) loadKV(kb + 1);

        const bool active = (kb * KB) <= (q0 + 15);
        if (active) {
            // ---- S = Q * K^T : two 16-key groups, K-dim = 128 in 4 slices ----
            v8f sc[2];
#pragma unroll
            for (int g = 0; g < 2; ++g) {
#pragma unroll
                for (int e = 0; e < 8; ++e) sc[g][e] = 0.0f;
                const unsigned short* krow = &Klds[g * 16 + lm][0];
#pragma unroll
                for (int ds = 0; ds < 4; ++ds) {
                    BfVec bk;
#pragma unroll
                    for (int j = 0; j < 8; ++j) {
                        const int d = ds * 32 + 16 * (j >> 2) + 8 * half + 2 * (j & 3);
                        bk.u32[j] = *(const unsigned int*)&krow[d];
                    }
                    sc[g] = __builtin_amdgcn_wmma_f32_16x16x32_bf16(
                        false, aq[ds].v, false, bk.v, (short)0, sc[g], false, false);
                }
            }

            // ---- causal mask + block row-max (scale already folded into Q) ----
            float pm[2][8], bmax[8];
#pragma unroll
            for (int r = 0; r < 8; ++r) {
                const int qrow = q0 + r + 8 * half;
                float t0 = sc[0][r];
                float t1 = sc[1][r];
                if (kb * KB + lm      > qrow) t0 = -1e30f;
                if (kb * KB + 16 + lm > qrow) t1 = -1e30f;
                pm[0][r] = t0; pm[1][r] = t1;
                bmax[r] = half16_max(fmaxf(t0, t1));
            }

            // ---- online max update + exponentials ----
            float alpha[8];
#pragma unroll
            for (int r = 0; r < 8; ++r) {
                const float mn = fmaxf(m_i[r], bmax[r]);
                alpha[r] = __expf(m_i[r] - mn);
                m_i[r] = mn;
                pm[0][r] = __expf(pm[0][r] - mn);
                pm[1][r] = __expf(pm[1][r] - mn);
            }

            // ---- P: C-layout -> LDS row-major -> A-layout (same-wave exchange) ----
#pragma unroll
            for (int g = 0; g < 2; ++g)
#pragma unroll
                for (int r = 0; r < 8; ++r)
                    Plds[wave][r + 8 * half][g * 16 + lm] = f2bf(pm[g][r]);

#pragma unroll
            for (int t = 0; t < 8; ++t)
#pragma unroll
                for (int r = 0; r < 8; ++r)
                    acc[t][r] *= alpha[r];

            BfVec ap;
            {
                const unsigned short* prow = &Plds[wave][lm][0];
#pragma unroll
                for (int p = 0; p < 8; ++p) {
                    const int kk = 16 * (p >> 2) + 8 * half + 2 * (p & 3);
                    ap.u32[p] = *(const unsigned int*)&prow[kk];
                }
            }

            // ---- row sums of P on the matrix core: rs = P * ones(32x16) ----
            v8f rs;
#pragma unroll
            for (int e = 0; e < 8; ++e) rs[e] = 0.0f;
            rs = __builtin_amdgcn_wmma_f32_16x16x32_bf16(
                false, ap.v, false, onesB.v, (short)0, rs, false, false);
#pragma unroll
            for (int r = 0; r < 8; ++r)
                l_i[r] = l_i[r] * alpha[r] + rs[r];

            // ---- O += P * V : 8 d-groups of 16 ----
#pragma unroll
            for (int t = 0; t < 8; ++t) {
                BfVec bv;
                const unsigned short* vtrow = &Vtlds[t * 16 + lm][0];
#pragma unroll
                for (int j = 0; j < 8; ++j)
                    bv.u32[j] = *(const unsigned int*)&vtrow[16 * (j >> 2) + 8 * half + 2 * (j & 3)];
                acc[t] = __builtin_amdgcn_wmma_f32_16x16x32_bf16(
                    false, ap.v, false, bv.v, (short)0, acc[t], false, false);
            }
        }
        __syncthreads();
    }

    // ---- epilogue: normalize and store fp32 ----
#pragma unroll
    for (int r = 0; r < 8; ++r) {
        const int qrow = q0 + r + 8 * half;
        float* op = out + (((size_t)b * S_ + (size_t)n * C_ + qrow) * H_ + h) * (size_t)D_;
        const float inv = 1.0f / l_i[r];
#pragma unroll
        for (int t = 0; t < 8; ++t)
            op[t * 16 + lm] = acc[t][r] * inv;
    }
}

extern "C" void kernel_launch(void* const* d_in, const int* in_sizes, int n_in,
                              void* d_out, int out_size, void* d_ws, size_t ws_size,
                              hipStream_t stream) {
    (void)in_sizes; (void)n_in; (void)d_ws; (void)ws_size; (void)out_size;
    const float* q     = (const float*)d_in[0];
    const float* k     = (const float*)d_in[1];
    const float* v     = (const float*)d_in[2];
    const float* sinks = (const float*)d_in[3];
    // d_in[4] = chunk_size (constant 1024, baked into the kernel)
    float* out = (float*)d_out;

    dim3 grid(B_ * H_ * NCHUNK * QT_PER_CHUNK);  // 2048 workgroups
    dim3 block(256);                             // 8 wave32 waves
    attn_sink_kernel<<<grid, block, 0, stream>>>(q, k, v, sinks, out);
}
